// FAENet_9887014715650
// MI455X (gfx1250) — matrix operand order repaired and checked by reading, not compiled, not run on hardware
//
#include <hip/hip_runtime.h>
#include <hip/hip_bf16.h>

typedef __attribute__((ext_vector_type(16))) _Float16 v16h;
typedef __attribute__((ext_vector_type(8)))  _Float16 v8h;
typedef __attribute__((ext_vector_type(8)))  float    v8f;

#define N_NODES  10000     // multiple of 16 -> no tile-tail handling
#define E_EDGES  400000    // multiple of 16 -> no tile-tail handling
#define G_GRAPHS 16
#define H_DIM    128
#define NG_GAUSS 50
#define L_LAYERS 4

static_assert(N_NODES % 16 == 0, "node tiles are exact");
static_assert(E_EDGES % 16 == 0, "edge tiles are exact");

// fast swish: x * sigmoid(x), sigmoid via v_exp_f32 + v_rcp_f32 (no IEEE divide)
__device__ __forceinline__ float swishf(float x) {
    return x * __builtin_amdgcn_rcpf(1.0f + __expf(-x));
}

// ---------------------------------------------------------------------------
// utility: zero a float buffer
// ---------------------------------------------------------------------------
__global__ void zero_f32_kernel(float* __restrict__ p, long long n) {
    long long i = (long long)blockIdx.x * blockDim.x + threadIdx.x;
    if (i < n) p[i] = 0.0f;
}

// ---------------------------------------------------------------------------
// Pre-swizzle a [K,128] f32 weight matrix into WMMA B-fragment order (f16):
// frag[((nt*KS + ks)*32 + lane)*16 + j] = W[ks*32 + (lane>>4)*8 + (j&7) + (j>>3)*16][nt*16 + lane&15]
// (CDNA5 ISA 16-bit B layout: lanes 0-15 hold K=0..7,16..23, lanes 16-31 K=8..15,24..31,
//  column N = lane%16). One contiguous 32-byte load per lane per fragment.
// ---------------------------------------------------------------------------
__global__ void swizzle_weight_kernel(const float* __restrict__ W,
                                      _Float16* __restrict__ frag, int KS) {
    int total = KS * 32 * 16 * 8;  // == K * 128
    for (int idx = blockIdx.x * blockDim.x + threadIdx.x; idx < total;
         idx += gridDim.x * blockDim.x) {
        int j    = idx & 15;
        int lane = (idx >> 4) & 31;
        int rem  = idx >> 9;
        int ks   = rem % KS;
        int nt   = rem / KS;
        int k    = ks * 32 + (lane >> 4) * 8 + (j & 7) + (j >> 3) * 16;
        int n    = nt * 16 + (lane & 15);
        frag[idx] = (_Float16)W[(size_t)k * 128 + n];
    }
}

// ---------------------------------------------------------------------------
// Node-feature gather: xin = concat(emb[z](14), tag[tags](32), phys[z](18),
//                                   period[pidx[z]](32), group[gidx[z]](32))
// ---------------------------------------------------------------------------
__global__ void gather_feats_kernel(const int* __restrict__ z, const int* __restrict__ tags,
                                    const int* __restrict__ pidx, const int* __restrict__ gidx,
                                    const float* __restrict__ emb, const float* __restrict__ tag_t,
                                    const float* __restrict__ phys, const float* __restrict__ per_t,
                                    const float* __restrict__ grp_t, _Float16* __restrict__ xin) {
    int i = blockIdx.x * blockDim.x + threadIdx.x;
    if (i >= N_NODES * 128) return;
    int n = i >> 7, c = i & 127;
    int zz = z[n];
    float v;
    if (c < 14)       v = emb[zz * 14 + c];
    else if (c < 46)  v = tag_t[tags[n] * 32 + (c - 14)];
    else if (c < 64)  v = phys[zz * 18 + (c - 46)];
    else if (c < 96)  v = per_t[pidx[zz] * 32 + (c - 64)];
    else              v = grp_t[gidx[zz] * 32 + (c - 96)];
    xin[i] = (_Float16)v;
}

// ---------------------------------------------------------------------------
// Edge embedding: e = swish([rel_pos@W1 | gauss(edge_weight)@W2] + b) -> f16
// one 128-thread block per edge
// ---------------------------------------------------------------------------
__global__ void edge_embed_kernel(const float* __restrict__ rel_pos,
                                  const float* __restrict__ ew,
                                  const float* __restrict__ W1, const float* __restrict__ b1,
                                  const float* __restrict__ W2, const float* __restrict__ b2,
                                  _Float16* __restrict__ e16) {
    int eidx = blockIdx.x;
    int t = threadIdx.x;
    __shared__ float attr[NG_GAUSS];
    __shared__ float rp[3];
    if (t < 3) rp[t] = rel_pos[(size_t)eidx * 3 + t];
    if (t < NG_GAUSS) {
        const float step  = 6.0f / 49.0f;
        const float coeff = -0.5f / (step * step);
        float d0 = ew[eidx] - (float)t * step;
        attr[t] = __expf(coeff * d0 * d0);
    }
    __syncthreads();
    float v;
    if (t < 64) {
        v = b1[t] + rp[0] * W1[t] + rp[1] * W1[64 + t] + rp[2] * W1[128 + t];
    } else {
        int c = t - 64;
        float acc = b2[c];
#pragma unroll 5
        for (int g = 0; g < NG_GAUSS; g++) acc += attr[g] * W2[g * 64 + c];
        v = acc;
    }
    e16[(size_t)eidx * 128 + t] = (_Float16)swishf(v);
}

// ---------------------------------------------------------------------------
// Generic node GEMM: y = swish(x[rows,128] @ W[128,128] + b)
// 256 threads = 8 waves; wave w owns n-tile w; 4 K-steps of v_wmma f16,
// A-fragment loads software-pipelined against the WMMA pipe.
// WRITE_F32 is a compile-time variant (no per-element branch in epilogue).
// ---------------------------------------------------------------------------
template <bool WRITE_F32>
__launch_bounds__(256)
__global__ void node_gemm_swish_kernel(const _Float16* __restrict__ x,
                                       const _Float16* __restrict__ wfrag,
                                       const float* __restrict__ bias,
                                       _Float16* __restrict__ y16,
                                       float* __restrict__ y32, int rows) {
    __shared__ _Float16 xt[16][136];  // 16 rows x 128 halves (+8 pad vs bank conflicts)
    int t = threadIdx.x;
    int wave = t >> 5, lane = t & 31;
    int hi = lane >> 4;                 // 0/1: which half of lanes
    int n = (wave << 4) + (lane & 15);  // output column
    float bn = bias[n];
    const _Float16* arow = &xt[lane & 15][hi * 8];
    int ntiles = rows >> 4;             // rows is a multiple of 16
    for (int tile = blockIdx.x; tile < ntiles; tile += gridDim.x) {
        {   // stage 16x128 f16 tile: 16 threads per row, 8 halves each
            int er = t >> 4, seg = t & 15;
            int r = tile * 16 + er;
            *(v8h*)&xt[er][seg * 8] = *(const v8h*)&x[(size_t)r * 128 + seg * 8];
        }
        __syncthreads();
        v8f acc = {};
        // A fragment per ISA 16-bit A layout: row M = lane%16,
        // halves 0..7  <- K = ks*32 + hi*8 .. +7 ; halves 8..15 <- +16
        v8h a0 = *(const v8h*)(arow);
        v8h a1 = *(const v8h*)(arow + 16);
#pragma unroll
        for (int ks = 0; ks < 4; ks++) {
            v16h a;
#pragma unroll
            for (int q = 0; q < 8; q++) { a[q] = a0[q]; a[q + 8] = a1[q]; }
            if (ks < 3) {   // pipeline next K-step's LDS loads over this WMMA
                a0 = *(const v8h*)(arow + (ks + 1) * 32);
                a1 = *(const v8h*)(arow + (ks + 1) * 32 + 16);
            }
            v16h b = *(const v16h*)&wfrag[(((size_t)wave * 4 + ks) * 32 + lane) << 4];
            acc = __builtin_amdgcn_wmma_f32_16x16x32_f16(false, a, false, b,
                                                         (short)0, acc, false, false);
        }
#pragma unroll
        for (int r8 = 0; r8 < 8; r8++) {        // C layout: M = r8 + hi*8, N = lane%16
            int row = tile * 16 + r8 + hi * 8;
            float v = swishf(acc[r8] + bn);
            y16[(size_t)row * 128 + n] = (_Float16)v;
            if (WRITE_F32) y32[(size_t)row * 128 + n] = v;
        }
        __syncthreads();
    }
}

// ---------------------------------------------------------------------------
// Interaction (the hot kernel): per 16-edge tile
//   ecat = [e | h[src] | h[dst]]   (16 x 384 f16, staged in LDS)
//   em   = swish(ecat @ geom_W + b)         (8 waves x 12 WMMA K-steps)
//   atomicAdd(agg[dst], em * h[src])
// Weights read as pre-swizzled B fragments straight from global (L2-resident).
// ---------------------------------------------------------------------------
__launch_bounds__(256)
__global__ void interaction_kernel(const _Float16* __restrict__ e16,
                                   const _Float16* __restrict__ h16,
                                   const _Float16* __restrict__ wfrag,
                                   const float* __restrict__ bias,
                                   const int* __restrict__ src,
                                   const int* __restrict__ dst,
                                   float* __restrict__ agg) {
    __shared__ _Float16 ecat[16][392];  // 384 + 8 pad (row stride 784B = 49x16B)
    __shared__ int sdst[16];
    int t = threadIdx.x;
    int wave = t >> 5, lane = t & 31;
    int hi = lane >> 4;
    int n = (wave << 4) + (lane & 15);
    float bn = bias[n];
    const _Float16* arow = &ecat[lane & 15][hi * 8];
    const int ntiles = E_EDGES >> 4;    // exact
    for (int tile = blockIdx.x; tile < ntiles; tile += gridDim.x) {
        {   // stage ecat: 16 threads per edge, 8 halves per segment per source
            int er = t >> 4, seg = t & 15;
            int eidx = tile * 16 + er;
            int s = src[eidx], d = dst[eidx];
            if (seg == 0) sdst[er] = d;
            *(v8h*)&ecat[er][seg * 8]       = *(const v8h*)&e16[(size_t)eidx * 128 + seg * 8];
            *(v8h*)&ecat[er][128 + seg * 8] = *(const v8h*)&h16[(size_t)s * 128 + seg * 8];
            *(v8h*)&ecat[er][256 + seg * 8] = *(const v8h*)&h16[(size_t)d * 128 + seg * 8];
            // prefetch next tile's e rows (streaming; lowers to global_prefetch_b8)
            int nx = tile + gridDim.x;
            if (nx < ntiles) {
                __builtin_prefetch(&e16[(size_t)(nx * 16 + er) * 128 + seg * 8], 0, 1);
            }
        }
        __syncthreads();
        v8f acc = {};
        v8h a0 = *(const v8h*)(arow);
        v8h a1 = *(const v8h*)(arow + 16);
#pragma unroll
        for (int ks = 0; ks < 12; ks++) {
            v16h a;
#pragma unroll
            for (int q = 0; q < 8; q++) { a[q] = a0[q]; a[q + 8] = a1[q]; }
            if (ks < 11) {  // pipeline next K-step's LDS loads over this WMMA
                a0 = *(const v8h*)(arow + (ks + 1) * 32);
                a1 = *(const v8h*)(arow + (ks + 1) * 32 + 16);
            }
            v16h b = *(const v16h*)&wfrag[(((size_t)wave * 12 + ks) * 32 + lane) << 4];
            acc = __builtin_amdgcn_wmma_f32_16x16x32_f16(false, a, false, b,
                                                         (short)0, acc, false, false);
        }
#pragma unroll
        for (int r8 = 0; r8 < 8; r8++) {
            int m = r8 + hi * 8;            // edge row within tile (C layout)
            float em = swishf(acc[r8] + bn);
            float hs = (float)ecat[m][128 + n];   // h[src] column n, from LDS
            atomicAdd(&agg[(size_t)sdst[m] * 128 + n], em * hs);
        }
        __syncthreads();
    }
}

// ---------------------------------------------------------------------------
// GraphNorm passes (batch is sorted -> local flush-accumulate minimizes atomics)
// ---------------------------------------------------------------------------
__global__ void gn_pass1_kernel(const float* __restrict__ agg, const int* __restrict__ batch,
                                float* __restrict__ gsum, float* __restrict__ counts) {
    int col = threadIdx.x;  // 128
    int start = blockIdx.x * 64;
    int end = start + 64; if (end > N_NODES) end = N_NODES;
    float acc = 0.f, cnt = 0.f; int curg = -1;
    for (int nd = start; nd < end; nd++) {
        int g = batch[nd];
        if (g != curg) {
            if (curg >= 0) {
                atomicAdd(&gsum[curg * 128 + col], acc);
                if (col == 0) atomicAdd(&counts[curg], cnt);
            }
            curg = g; acc = 0.f; cnt = 0.f;
        }
        acc += agg[(size_t)nd * 128 + col];
        cnt += 1.f;
    }
    if (curg >= 0) {
        atomicAdd(&gsum[curg * 128 + col], acc);
        if (col == 0) atomicAdd(&counts[curg], cnt);
    }
}

__global__ void gn_pass2_kernel(const float* __restrict__ agg, const int* __restrict__ batch,
                                const float* __restrict__ gsum, const float* __restrict__ counts,
                                const float* __restrict__ ms,
                                float* __restrict__ gout, float* __restrict__ gvar) {
    int col = threadIdx.x;
    int start = blockIdx.x * 64;
    int end = start + 64; if (end > N_NODES) end = N_NODES;
    float msv = ms[col];
    float acc = 0.f, mean = 0.f; int curg = -1;
    for (int nd = start; nd < end; nd++) {
        int g = batch[nd];
        if (g != curg) {
            if (curg >= 0) atomicAdd(&gvar[curg * 128 + col], acc);
            curg = g; acc = 0.f;
            mean = gsum[g * 128 + col] / fmaxf(counts[g], 1.f);
        }
        float o = agg[(size_t)nd * 128 + col] - mean * msv;
        gout[(size_t)nd * 128 + col] = o;
        acc += o * o;
    }
    if (curg >= 0) atomicAdd(&gvar[curg * 128 + col], acc);
}

__global__ void gn_pass3_kernel(const float* __restrict__ gout, const int* __restrict__ batch,
                                const float* __restrict__ gvar, const float* __restrict__ counts,
                                const float* __restrict__ w, const float* __restrict__ b,
                                _Float16* __restrict__ hh16) {
    int i = blockIdx.x * blockDim.x + threadIdx.x;
    if (i >= N_NODES * 128) return;
    int nd = i >> 7, col = i & 127;
    int g = batch[nd];
    float var = gvar[g * 128 + col] / fmaxf(counts[g], 1.f);
    float y = w[col] * gout[i] * rsqrtf(var + 1e-5f) + b[col];
    hh16[i] = (_Float16)swishf(y);
}

// ---------------------------------------------------------------------------
// Output block: per node, alpha = h@w_lin + wb; t = (swish(h@W1+b1)@W2 + b2)*alpha;
// segment-sum into energies[g*(L+1)+li]. One 128-thread block per node.
// ---------------------------------------------------------------------------
__global__ void output_block_kernel(const _Float16* __restrict__ h16,
                                    const float* __restrict__ W1, const float* __restrict__ b1,
                                    const float* __restrict__ W2, const float* __restrict__ b2,
                                    const float* __restrict__ wW, const float* __restrict__ wb,
                                    const int* __restrict__ batch,
                                    float* __restrict__ energies, int li) {
    int node = blockIdx.x;
    int t = threadIdx.x;  // 128
    __shared__ float hsh[128];
    __shared__ float red[128];
    float hv = (float)h16[(size_t)node * 128 + t];
    hsh[t] = hv;
    red[t] = hv * wW[t];
    __syncthreads();
    for (int s = 64; s > 0; s >>= 1) { if (t < s) red[t] += red[t + s]; __syncthreads(); }
    float alpha = red[0] + wb[0];
    __syncthreads();
    float tv = 0.f;
    if (t < 64) {
        float acc = b1[t];
#pragma unroll 8
        for (int k = 0; k < 128; k++) acc += hsh[k] * W1[k * 64 + t];
        tv = swishf(acc) * W2[t];
    }
    red[t] = (t < 64) ? tv : 0.f;
    __syncthreads();
    for (int s = 64; s > 0; s >>= 1) { if (t < s) red[t] += red[t + s]; __syncthreads(); }
    if (t == 0) {
        float val = (red[0] + b2[0]) * alpha;
        atomicAdd(&energies[batch[node] * (L_LAYERS + 1) + li], val);
    }
}

// ---------------------------------------------------------------------------
// Finalize: energy[g] = energies[g,:]@skip_W + skip_b ; then copy h (f32)
// ---------------------------------------------------------------------------
__global__ void finalize_kernel(const float* __restrict__ energies,
                                const float* __restrict__ skipW, const float* __restrict__ skipb,
                                const float* __restrict__ h32, float* __restrict__ out) {
    long long idx = (long long)blockIdx.x * blockDim.x + threadIdx.x;
    long long total = 16 + (long long)N_NODES * 128;
    if (idx >= total) return;
    if (idx < 16) {
        float s = skipb[0];
#pragma unroll
        for (int j = 0; j < L_LAYERS + 1; j++) s += energies[idx * (L_LAYERS + 1) + j] * skipW[j];
        out[idx] = s;
    } else {
        out[idx] = h32[idx - 16];
    }
}

// ---------------------------------------------------------------------------
extern "C" void kernel_launch(void* const* d_in, const int* in_sizes, int n_in,
                              void* d_out, int out_size, void* d_ws, size_t ws_size,
                              hipStream_t stream) {
    const float* rel_pos     = (const float*)d_in[0];
    const float* edge_weight = (const float*)d_in[1];
    const float* phys        = (const float*)d_in[2];
    const float* emb         = (const float*)d_in[3];
    const float* tag_t       = (const float*)d_in[4];
    const float* per_t       = (const float*)d_in[5];
    const float* grp_t       = (const float*)d_in[6];
    const float* lin_W       = (const float*)d_in[7];
    const float* lin_b       = (const float*)d_in[8];
    const float* e1_W        = (const float*)d_in[9];
    const float* e1_b        = (const float*)d_in[10];
    const float* e12_W       = (const float*)d_in[11];
    const float* e12_b       = (const float*)d_in[12];
    const float* geom_W      = (const float*)d_in[13];
    const float* geom_b      = (const float*)d_in[14];
    const float* linh_W      = (const float*)d_in[15];
    const float* linh_b      = (const float*)d_in[16];
    const float* mlp_W       = (const float*)d_in[17];
    const float* mlp_b       = (const float*)d_in[18];
    const float* gn_w        = (const float*)d_in[19];
    const float* gn_b        = (const float*)d_in[20];
    const float* gn_ms       = (const float*)d_in[21];
    const float* ol1_W       = (const float*)d_in[22];
    const float* ol1_b       = (const float*)d_in[23];
    const float* ol2_W       = (const float*)d_in[24];
    const float* ol2_b       = (const float*)d_in[25];
    const float* wl_W        = (const float*)d_in[26];
    const float* wl_b        = (const float*)d_in[27];
    const float* skip_W      = (const float*)d_in[28];
    const float* skip_b      = (const float*)d_in[29];
    const int*   z           = (const int*)d_in[30];
    const int*   tags        = (const int*)d_in[31];
    const int*   pidx        = (const int*)d_in[32];
    const int*   gidx        = (const int*)d_in[33];
    const int*   eindex      = (const int*)d_in[34];
    const int*   batch       = (const int*)d_in[35];
    const int*   src         = eindex;
    const int*   dst         = eindex + E_EDGES;

    // --- carve workspace (no allocation; graph-capture safe) ---
    char* p = (char*)d_ws;
    auto carve = [&](size_t bytes) -> char* {
        char* r = p; p += (bytes + 255) & ~(size_t)255; return r;
    };
    const size_t NH = (size_t)N_NODES * 128;
    _Float16* e16    = (_Float16*)carve((size_t)E_EDGES * 128 * 2);
    _Float16* xin16  = (_Float16*)carve(NH * 2);
    _Float16* h16    = (_Float16*)carve(NH * 2);
    _Float16* hh16   = (_Float16*)carve(NH * 2);
    _Float16* t16    = (_Float16*)carve(NH * 2);
    float*    h32    = (float*)carve(NH * 4);
    float*    agg    = (float*)carve(NH * 4);
    float*    gout   = (float*)carve(NH * 4);
    _Float16* frag_lin  = (_Float16*)carve((size_t)128 * 128 * 2);
    _Float16* frag_geom = (_Float16*)carve((size_t)L_LAYERS * 384 * 128 * 2);
    _Float16* frag_linh = (_Float16*)carve((size_t)L_LAYERS * 128 * 128 * 2);
    _Float16* frag_mlp  = (_Float16*)carve((size_t)L_LAYERS * 128 * 128 * 2);
    float*    gsum   = (float*)carve(G_GRAPHS * 128 * 4);
    float*    gvar   = (float*)carve(G_GRAPHS * 128 * 4);
    float*    counts = (float*)carve(G_GRAPHS * 4);
    float*    energ  = (float*)carve(G_GRAPHS * (L_LAYERS + 1) * 4);

    const int gnBlocks   = (N_NODES + 63) / 64;
    const int tilesNode  = N_NODES / 16;

    // zero accumulators (per-call: deterministic)
    zero_f32_kernel<<<1, 256, 0, stream>>>(energ, G_GRAPHS * (L_LAYERS + 1));

    // weight pre-swizzle into WMMA B-fragment order
    swizzle_weight_kernel<<<64, 256, 0, stream>>>(lin_W, frag_lin, 4);
    for (int i = 0; i < L_LAYERS; i++) {
        swizzle_weight_kernel<<<192, 256, 0, stream>>>(geom_W + (size_t)i * 384 * 128,
                                                       frag_geom + (size_t)i * 384 * 128, 12);
        swizzle_weight_kernel<<<64, 256, 0, stream>>>(linh_W + (size_t)i * 128 * 128,
                                                      frag_linh + (size_t)i * 128 * 128, 4);
        swizzle_weight_kernel<<<64, 256, 0, stream>>>(mlp_W + (size_t)i * 128 * 128,
                                                      frag_mlp + (size_t)i * 128 * 128, 4);
    }

    // embeddings
    gather_feats_kernel<<<(N_NODES * 128 + 255) / 256, 256, 0, stream>>>(
        z, tags, pidx, gidx, emb, tag_t, phys, per_t, grp_t, xin16);
    edge_embed_kernel<<<E_EDGES, 128, 0, stream>>>(rel_pos, edge_weight,
                                                   e1_W, e1_b, e12_W, e12_b, e16);
    node_gemm_swish_kernel<false><<<tilesNode, 256, 0, stream>>>(
        xin16, frag_lin, lin_b, h16, (float*)nullptr, N_NODES);

    // interaction layers
    for (int i = 0; i < L_LAYERS; i++) {
        output_block_kernel<<<N_NODES, 128, 0, stream>>>(h16, ol1_W, ol1_b, ol2_W, ol2_b,
                                                         wl_W, wl_b, batch, energ, i);
        zero_f32_kernel<<<(int)((NH + 255) / 256), 256, 0, stream>>>(agg, (long long)NH);
        interaction_kernel<<<2048, 256, 0, stream>>>(e16, h16,
                                                     frag_geom + (size_t)i * 384 * 128,
                                                     geom_b + (size_t)i * 128,
                                                     src, dst, agg);
        zero_f32_kernel<<<1, 256, 0, stream>>>(gsum, G_GRAPHS * 128);
        zero_f32_kernel<<<1, 256, 0, stream>>>(gvar, G_GRAPHS * 128);
        zero_f32_kernel<<<1, 256, 0, stream>>>(counts, G_GRAPHS);
        gn_pass1_kernel<<<gnBlocks, 128, 0, stream>>>(agg, batch, gsum, counts);
        gn_pass2_kernel<<<gnBlocks, 128, 0, stream>>>(agg, batch, gsum, counts,
                                                      gn_ms + (size_t)i * 128, gout, gvar);
        gn_pass3_kernel<<<(N_NODES * 128 + 255) / 256, 256, 0, stream>>>(
            gout, batch, gvar, counts, gn_w + (size_t)i * 128, gn_b + (size_t)i * 128, hh16);
        node_gemm_swish_kernel<false><<<tilesNode, 256, 0, stream>>>(
            hh16, frag_linh + (size_t)i * 128 * 128, linh_b + (size_t)i * 128,
            t16, (float*)nullptr, N_NODES);
        if (i == L_LAYERS - 1) {
            node_gemm_swish_kernel<true><<<tilesNode, 256, 0, stream>>>(
                t16, frag_mlp + (size_t)i * 128 * 128, mlp_b + (size_t)i * 128,
                h16, h32, N_NODES);
        } else {
            node_gemm_swish_kernel<false><<<tilesNode, 256, 0, stream>>>(
                t16, frag_mlp + (size_t)i * 128 * 128, mlp_b + (size_t)i * 128,
                h16, (float*)nullptr, N_NODES);
        }
    }
    output_block_kernel<<<N_NODES, 128, 0, stream>>>(h16, ol1_W, ol1_b, ol2_W, ol2_b,
                                                     wl_W, wl_b, batch, energ, L_LAYERS);

    long long total = 16 + (long long)N_NODES * 128;
    finalize_kernel<<<(int)((total + 255) / 256), 256, 0, stream>>>(
        energ, skip_W, skip_b, h32, (float*)d_out);
}
